// QuantizedLinear_8143257994103
// MI455X (gfx1250) — matrix operand order repaired and checked by reading, not compiled, not run on hardware
//
#include <hip/hip_runtime.h>
#include <hip/hip_bf16.h>

// ---- types ----
typedef __bf16 bf16_t;
typedef __attribute__((ext_vector_type(16))) __bf16 v16bf;
typedef __attribute__((ext_vector_type(8)))  __bf16 v8bf;
typedef __attribute__((ext_vector_type(8)))  float  v8f;
typedef __attribute__((ext_vector_type(4)))  float  v4f;

#define M_DIM 4096
#define K_DIM 4096
#define N_DIM 16384

// ================= async-to-LDS support (CDNA5) =================
#if defined(__AMDGCN__) && __has_builtin(__builtin_amdgcn_global_load_async_to_lds_b128)
#define HAVE_ASYNC_LDS 1
// builtin signature (from hipcc diagnostic): param0 = int4 __device__* (AS1),
// param1 = int4 LDS* (AS3), then two immediate ints (offset, cpol)
typedef int v4i_gcc __attribute__((vector_size(16)));
typedef v4i_gcc __attribute__((address_space(1)))* g_v4i_ptr;
typedef v4i_gcc __attribute__((address_space(3)))* l_v4i_ptr;
__device__ __forceinline__ void async_copy16(const void* g, void* l) {
    __builtin_amdgcn_global_load_async_to_lds_b128(
        (g_v4i_ptr)(v4i_gcc*)const_cast<void*>(g),
        (l_v4i_ptr)(v4i_gcc*)l, 0, 0);
}
#else
#define HAVE_ASYNC_LDS 0
#endif

__device__ __forceinline__ void wait_async_zero() {
#if defined(__AMDGCN__)
#if __has_builtin(__builtin_amdgcn_s_wait_asynccnt)
    __builtin_amdgcn_s_wait_asynccnt(0);
#else
    asm volatile("s_wait_asynccnt 0x0" ::: "memory");
#endif
#endif
}

// =================================================================
// Prepass 1: x f32 -> bf16   (bandwidth bound, ~4us)
// =================================================================
__global__ __launch_bounds__(256) void convert_x_bf16(
    const float* __restrict__ x, bf16_t* __restrict__ xb)
{
    size_t i = ((size_t)blockIdx.x * 256 + threadIdx.x) * 8;
    v4f a = *(const v4f*)(x + i);
    v4f b = *(const v4f*)(x + i + 4);
    v8bf p;
    #pragma unroll
    for (int j = 0; j < 4; ++j) { p[j] = (bf16_t)a[j]; p[4 + j] = (bf16_t)b[j]; }
    *(v8bf*)(xb + i) = p;
}

// =================================================================
// Prepass 2: w int8 [N][K] -> bf16 [K][N] (transpose via LDS, ~8us)
// =================================================================
#define TWN 64
#define TWK 64
#define TWP 72   // padded K stride in tile (144B rows, 16B aligned)
__global__ __launch_bounds__(256) void convert_w_bf16_t(
    const signed char* __restrict__ w, bf16_t* __restrict__ wt)
{
    __shared__ bf16_t tile[TWN * TWP];
    const int nBase = blockIdx.x * TWN;
    const int kBase = blockIdx.y * TWK;
    const int t = threadIdx.x;

    {   // load: 16 contiguous int8 along K for one N row, convert
        const int n  = t >> 2;
        const int k0 = (t & 3) * 16;
        int4 raw = *(const int4*)(w + (size_t)(nBase + n) * K_DIM + kBase + k0);
        const signed char* rc = (const signed char*)&raw;
        v8bf q0, q1;
        #pragma unroll
        for (int j = 0; j < 8; ++j) {
            q0[j] = (bf16_t)(float)(int)rc[j];
            q1[j] = (bf16_t)(float)(int)rc[8 + j];
        }
        *(v8bf*)&tile[n * TWP + k0]     = q0;
        *(v8bf*)&tile[n * TWP + k0 + 8] = q1;
    }
    __syncthreads();
    {   // store transposed: 16 contiguous N for one K row
        const int k  = t >> 2;
        const int n0 = (t & 3) * 16;
        v8bf q0, q1;
        #pragma unroll
        for (int j = 0; j < 8; ++j) {
            q0[j] = tile[(n0 + j) * TWP + k];
            q1[j] = tile[(n0 + 8 + j) * TWP + k];
        }
        bf16_t* dst = wt + (size_t)(kBase + k) * N_DIM + nBase + n0;
        *(v8bf*)dst       = q0;
        *(v8bf*)(dst + 8) = q1;
    }
}

// =================================================================
// Main GEMM: pure bf16, double-buffered LDS, async-to-LDS staging
// Block tile 128x128, BK=64. 8 waves: 4(M) x 2(N); wave tile 32x64.
// =================================================================
#define BM 128
#define BN 128
#define BK 64
#define LDA 72    // bf16 stride of As rows (144B, 16B aligned)
#define LDB 136   // bf16 stride of Bs rows (272B, 16B aligned)

__global__ __launch_bounds__(256) void qlinear_gemm_bf16(
    const bf16_t* __restrict__ xb,     // [M][K] bf16
    const bf16_t* __restrict__ wt,     // [K][N] bf16 (int values of weight)
    const float*  __restrict__ scale,  // [N]
    const float*  __restrict__ bias,   // [N]
    float* __restrict__ out)           // [M][N]
{
    __shared__ bf16_t As[2][BM * LDA];
    __shared__ bf16_t Bs[2][BK * LDB];

    const int tid    = threadIdx.x;
    const int lane   = tid & 31;
    const int wave   = tid >> 5;
    const int laneLo = lane & 15;
    const int laneHi = lane >> 4;

    const int blockM = blockIdx.y * BM;
    const int blockN = blockIdx.x * BN;
    const int waveM  = (wave & 3) * 32;
    const int waveN  = (wave >> 2) * 64;

    v8f acc[2][4] = {};

    // staging maps: 4 x b128 per thread for each of A and B
    const int aRow = tid >> 1;              // 0..127
    const int aCol = (tid & 1) * 32;        // 0 or 32
    const int bK   = tid >> 2;              // 0..63
    const int bCol = (tid & 3) * 32;        // 0,32,64,96

    const bf16_t* aG = xb + (size_t)(blockM + aRow) * K_DIM + aCol;
    const bf16_t* bG = wt + (size_t)bK * N_DIM + blockN + bCol;

    bf16_t* aL = &As[0][aRow * LDA + aCol];
    bf16_t* bL = &Bs[0][bK * LDB + bCol];
    const size_t lsHalfA = BM * LDA;   // elements between buffers
    const size_t lsHalfB = BK * LDB;

    auto stage = [&](int buf, int kk) {
        const bf16_t* ag = aG + kk;
        const bf16_t* bg = bG + (size_t)kk * N_DIM;
        bf16_t* al = aL + (size_t)buf * lsHalfA;
        bf16_t* bl = bL + (size_t)buf * lsHalfB;
#if HAVE_ASYNC_LDS
        #pragma unroll
        for (int u = 0; u < 4; ++u) async_copy16(ag + u * 8, al + u * 8);
        #pragma unroll
        for (int u = 0; u < 4; ++u) async_copy16(bg + u * 8, bl + u * 8);
#else
        v8bf ra[4], rb[4];
        #pragma unroll
        for (int u = 0; u < 4; ++u) ra[u] = *(const v8bf*)(ag + u * 8);
        #pragma unroll
        for (int u = 0; u < 4; ++u) rb[u] = *(const v8bf*)(bg + u * 8);
        #pragma unroll
        for (int u = 0; u < 4; ++u) *(v8bf*)(al + u * 8) = ra[u];
        #pragma unroll
        for (int u = 0; u < 4; ++u) *(v8bf*)(bl + u * 8) = rb[u];
#endif
    };

    auto compute = [&](int buf) {
        const bf16_t* as = &As[buf][0];
        const bf16_t* bs = &Bs[buf][0];
        #pragma unroll
        for (int ks = 0; ks < BK; ks += 32) {
            #pragma unroll
            for (int mt = 0; mt < 2; ++mt) {
                const int m = waveM + mt * 16 + laneLo;
                v8bf alo = *(const v8bf*)&as[m * LDA + ks + laneHi * 8];
                v8bf ahi = *(const v8bf*)&as[m * LDA + ks + 16 + laneHi * 8];
                v16bf afrag = __builtin_shufflevector(alo, ahi,
                    0,1,2,3,4,5,6,7,8,9,10,11,12,13,14,15);
                #pragma unroll
                for (int nt = 0; nt < 4; ++nt) {
                    const int nb = waveN + nt * 16;
                    v16bf bfrag;
                    #pragma unroll
                    for (int s = 0; s < 16; ++s)
                        bfrag[s] = bs[(ks + (s < 8 ? laneHi * 8 + s : 16 + laneHi * 8 + (s - 8))) * LDB + nb + laneLo];
                    acc[mt][nt] = __builtin_amdgcn_wmma_f32_16x16x32_bf16(
                        false, afrag, false, bfrag,
                        (short)0, acc[mt][nt], false, false);
                }
            }
        }
    };

    // pipeline: prefill buf0, then double-buffer
    stage(0, 0);
    wait_async_zero();
    __syncthreads();

    int buf = 0;
    for (int kk = 0; kk < K_DIM; kk += BK) {
        const int nxt = buf ^ 1;
        if (kk + BK < K_DIM) stage(nxt, kk + BK);
        compute(buf);
        wait_async_zero();
        __syncthreads();
        buf = nxt;
    }

    // epilogue: out = acc * scale[n] + bias[n]
    #pragma unroll
    for (int nt = 0; nt < 4; ++nt) {
        const int n = blockN + waveN + nt * 16 + laneLo;
        const float sc = scale[n];
        const float bv = bias[n];
        #pragma unroll
        for (int mt = 0; mt < 2; ++mt) {
            const int mBase = blockM + waveM + mt * 16 + laneHi * 8;
            #pragma unroll
            for (int j = 0; j < 8; ++j) {
                out[(size_t)(mBase + j) * N_DIM + n] = acc[mt][nt][j] * sc + bv;
            }
        }
    }
}

// =================================================================
// Fallback fused kernel (round-1 version) if workspace too small
// =================================================================
#define FLDA 40
#define FLDB 136
__global__ __launch_bounds__(256) void qlinear_fused_bf16(
    const float* __restrict__ x,
    const signed char* __restrict__ w,
    const float* __restrict__ scale,
    const float* __restrict__ bias,
    float* __restrict__ out)
{
    __shared__ bf16_t As[BM * FLDA];
    __shared__ bf16_t Bs[32 * FLDB];

    const int tid    = threadIdx.x;
    const int lane   = tid & 31;
    const int wave   = tid >> 5;
    const int laneLo = lane & 15;
    const int laneHi = lane >> 4;

    const int blockM = blockIdx.y * BM;
    const int blockN = blockIdx.x * BN;
    const int waveM  = (wave & 3) * 32;
    const int waveN  = (wave >> 2) * 64;

    v8f acc[2][4] = {};

    const int aRow = tid >> 1;
    const int aCol = (tid & 1) * 16;
    const float* xRow = x + (size_t)(blockM + aRow) * K_DIM + aCol;

    const int bN  = tid & 127;
    const int bK0 = (tid >> 7) * 16;
    const signed char* wRow = w + (size_t)(blockN + bN) * K_DIM + bK0;

    for (int kk = 0; kk < K_DIM; kk += 32) {
        v4f a0 = *(const v4f*)(xRow + kk + 0);
        v4f a1 = *(const v4f*)(xRow + kk + 4);
        v4f a2 = *(const v4f*)(xRow + kk + 8);
        v4f a3 = *(const v4f*)(xRow + kk + 12);
        v8bf p0, p1;
        #pragma unroll
        for (int i = 0; i < 4; ++i) {
            p0[i] = (bf16_t)a0[i]; p0[4 + i] = (bf16_t)a1[i];
            p1[i] = (bf16_t)a2[i]; p1[4 + i] = (bf16_t)a3[i];
        }
        bf16_t* asDst = &As[aRow * FLDA + aCol];
        *(v8bf*)(asDst + 0) = p0;
        *(v8bf*)(asDst + 8) = p1;

        int4 wb = *(const int4*)(wRow + kk);
        const signed char* wc = (const signed char*)&wb;
        #pragma unroll
        for (int j = 0; j < 16; ++j)
            Bs[(bK0 + j) * FLDB + bN] = (bf16_t)(float)(int)wc[j];

        __syncthreads();

        #pragma unroll
        for (int mt = 0; mt < 2; ++mt) {
            const int m = waveM + mt * 16 + laneLo;
            v8bf alo = *(const v8bf*)&As[m * FLDA + laneHi * 8];
            v8bf ahi = *(const v8bf*)&As[m * FLDA + 16 + laneHi * 8];
            v16bf afrag = __builtin_shufflevector(alo, ahi,
                0,1,2,3,4,5,6,7,8,9,10,11,12,13,14,15);
            #pragma unroll
            for (int nt = 0; nt < 4; ++nt) {
                const int nb = waveN + nt * 16;
                v16bf bfrag;
                #pragma unroll
                for (int s = 0; s < 16; ++s)
                    bfrag[s] = Bs[(s < 8 ? laneHi * 8 + s : 16 + laneHi * 8 + (s - 8)) * FLDB + nb + laneLo];
                acc[mt][nt] = __builtin_amdgcn_wmma_f32_16x16x32_bf16(
                    false, afrag, false, bfrag, (short)0, acc[mt][nt], false, false);
            }
        }
        __syncthreads();
    }

    #pragma unroll
    for (int nt = 0; nt < 4; ++nt) {
        const int n = blockN + waveN + nt * 16 + laneLo;
        const float sc = scale[n];
        const float bv = bias[n];
        #pragma unroll
        for (int mt = 0; mt < 2; ++mt) {
            const int mBase = blockM + waveM + mt * 16 + laneHi * 8;
            #pragma unroll
            for (int j = 0; j < 8; ++j) {
                out[(size_t)(mBase + j) * N_DIM + n] = acc[mt][nt][j] * sc + bv;
            }
        }
    }
}

// =================================================================
extern "C" void kernel_launch(void* const* d_in, const int* in_sizes, int n_in,
                              void* d_out, int out_size, void* d_ws, size_t ws_size,
                              hipStream_t stream) {
    (void)in_sizes; (void)n_in; (void)out_size;
    const float*       x     = (const float*)d_in[0];
    const signed char* wgt   = (const signed char*)d_in[1];
    const float*       scale = (const float*)d_in[2];
    const float*       bias  = (const float*)d_in[3];
    float*             out   = (float*)d_out;

    const size_t xbBytes = (size_t)M_DIM * K_DIM * sizeof(bf16_t);   // 32 MB
    const size_t wtBytes = (size_t)K_DIM * N_DIM * sizeof(bf16_t);   // 128 MB

    if (ws_size >= xbBytes + wtBytes) {
        bf16_t* xb = (bf16_t*)d_ws;
        bf16_t* wt = (bf16_t*)((char*)d_ws + xbBytes);
        convert_x_bf16<<<(M_DIM * (size_t)K_DIM) / (256 * 8), 256, 0, stream>>>(x, xb);
        convert_w_bf16_t<<<dim3(N_DIM / TWN, K_DIM / TWK), 256, 0, stream>>>(wgt, wt);
        qlinear_gemm_bf16<<<dim3(N_DIM / BN, M_DIM / BM), 256, 0, stream>>>(
            xb, wt, scale, bias, out);
    } else {
        qlinear_fused_bf16<<<dim3(N_DIM / BN, M_DIM / BM), 256, 0, stream>>>(
            x, wgt, scale, bias, out);
    }
}
